// NysAttention_59528246722948
// MI455X (gfx1250) — matrix operand order repaired and checked by reading, not compiled, not run on hardware
//
#include <hip/hip_runtime.h>
#include <hip/hip_fp16.h>

// Nystromformer attention forward for gfx1250 (MI455X), WMMA f16->f32 path.
// B=4, N=4096, C=768, H=12, hd=64, Lm=64.

typedef __attribute__((ext_vector_type(16))) _Float16 v16h;
typedef __attribute__((ext_vector_type(8)))  _Float16 v8h;
typedef __attribute__((ext_vector_type(8)))  float    v8f;

#define H_    12
#define N_    4096
#define C_    768
#define HD    64
#define LM    64
#define BH    48      // B * H
#define MROWS 16384   // B * N
#define C3    2304    // 3 * C

// ---------------- WMMA core: one wave computes a 32x32 tile of A @ B^T ----------------
// A: [M, K] row-major (lda = K stride), B: [Ncols, K] row-major (ldb), both f16.
// A-fragment layout (16-bit A 16x32): lanes 0-15 hold K 0..7 & 16..23, lanes 16-31 hold K 8..15 & 24..31.
// B-fragment layout (16-bit B 32x16): lanes 0-15 hold K 0..7 & 16..23, lanes 16-31 hold K 8..15 & 24..31 (col = lane&15).

__device__ __forceinline__ v16h load_a_frag(const _Float16* A, int lda, int row, int kb, int hf) {
  const _Float16* p0 = A + (size_t)row * lda + kb + hf * 8;
  v8h lo = *(const v8h*)p0;
  v8h hi = *(const v8h*)(p0 + 16);
  return __builtin_shufflevector(lo, hi, 0,1,2,3,4,5,6,7,8,9,10,11,12,13,14,15);
}

__device__ __forceinline__ v16h load_b_frag(const _Float16* B, int ldb, int col, int kb, int hf) {
  const _Float16* p0 = B + (size_t)col * ldb + kb + hf * 8;
  v8h lo = *(const v8h*)p0;
  v8h hi = *(const v8h*)(p0 + 16);
  return __builtin_shufflevector(lo, hi, 0,1,2,3,4,5,6,7,8,9,10,11,12,13,14,15);
}

__device__ __forceinline__ void wmma_tile32(const _Float16* A, int lda,
                                            const _Float16* B, int ldb,
                                            int K, int m0, int n0,
                                            int lane, v8f acc[2][2]) {
  const int hf = lane >> 4;
  const int r  = lane & 15;
  for (int kb = 0; kb < K; kb += 32) {
    v16h a0 = load_a_frag(A, lda, m0 + r,      kb, hf);
    v16h a1 = load_a_frag(A, lda, m0 + 16 + r, kb, hf);
    v16h b0 = load_b_frag(B, ldb, n0 + r,      kb, hf);
    v16h b1 = load_b_frag(B, ldb, n0 + 16 + r, kb, hf);
    acc[0][0] = __builtin_amdgcn_wmma_f32_16x16x32_f16(false, a0, false, b0, (short)0, acc[0][0], false, false);
    acc[0][1] = __builtin_amdgcn_wmma_f32_16x16x32_f16(false, a0, false, b1, (short)0, acc[0][1], false, false);
    acc[1][0] = __builtin_amdgcn_wmma_f32_16x16x32_f16(false, a1, false, b0, (short)0, acc[1][0], false, false);
    acc[1][1] = __builtin_amdgcn_wmma_f32_16x16x32_f16(false, a1, false, b1, (short)0, acc[1][1], false, false);
  }
}

// C/D layout: vgpr element e, lanes 0-15: M = e, N = lane; lanes 16-31: M = 8+e, N = lane-16.
template <typename F>
__device__ __forceinline__ void epilogue(int m0, int n0, int lane, const v8f acc[2][2], F f) {
  const int hf = lane >> 4, r = lane & 15;
  for (int mt = 0; mt < 2; ++mt)
    for (int nt = 0; nt < 2; ++nt)
      for (int e = 0; e < 8; ++e) {
        const int m = m0 + mt * 16 + hf * 8 + e;
        const int n = n0 + nt * 16 + r;
        f(m, n, acc[mt][nt][e]);
      }
}

// ---------------- converts ----------------
__global__ void k_cvt(const float* __restrict__ in, _Float16* __restrict__ out, int n) {
  int i = blockIdx.x * blockDim.x + threadIdx.x;
  if (i < n) out[i] = (_Float16)in[i];
}

// ---------------- QKV projection: qkv = x @ w_qkv^T, scatter to head layout ----------------
// q scaled by 1/8 (hd^-0.5), v stored transposed: vt[bh][d][n].
__global__ void k_gemm_qkv(const _Float16* __restrict__ xh, const _Float16* __restrict__ wh,
                           _Float16* __restrict__ qh, _Float16* __restrict__ kh,
                           _Float16* __restrict__ vt) {
  const int lane = threadIdx.x & 31;
  const int n0 = blockIdx.x * 32;   // over 2304
  const int m0 = blockIdx.y * 32;   // over 16384
  v8f acc[2][2] = {};
  wmma_tile32(xh, C_, wh, C_, C_, m0, n0, lane, acc);
  epilogue(m0, n0, lane, acc, [&](int m, int n, float val) {
    const int b = m >> 12;
    const int row = m & 4095;
    const int which = n / C_;
    const int rem = n % C_;
    const int h = rem >> 6;
    const int d = rem & 63;
    const int bh = b * H_ + h;
    if (which == 0)
      qh[((size_t)bh * N_ + row) * HD + d] = (_Float16)(val * 0.125f);
    else if (which == 1)
      kh[((size_t)bh * N_ + row) * HD + d] = (_Float16)val;
    else
      vt[((size_t)bh * HD + d) * N_ + row] = (_Float16)val;
  });
}

// ---------------- landmark pooling: mean over 64 contiguous rows ----------------
__global__ void k_pool(const _Float16* __restrict__ qh, const _Float16* __restrict__ kh,
                       _Float16* __restrict__ qlh, _Float16* __restrict__ klh) {
  const int idx = blockIdx.x;           // bh*64 + l
  const int bh = idx >> 6;
  const int l = idx & 63;
  const int d = threadIdx.x;            // 64 threads
  const size_t base = ((size_t)bh * N_ + (size_t)l * 64) * HD + d;
  float sq = 0.f, sk = 0.f;
  for (int s = 0; s < 64; ++s) {
    sq += (float)qh[base + (size_t)s * HD];
    sk += (float)kh[base + (size_t)s * HD];
  }
  const size_t o = ((size_t)bh * LM + l) * HD + d;
  qlh[o] = (_Float16)(sq * (1.f / 64.f));
  klh[o] = (_Float16)(sk * (1.f / 64.f));
}

// ---------------- batched GEMM (per-head) with f32 output ----------------
__global__ void k_gemm_f32out(const _Float16* __restrict__ A, int lda, size_t sA,
                              const _Float16* __restrict__ B, int ldb, size_t sB,
                              float* __restrict__ C, int ldc, size_t sC, int K) {
  const int bh = blockIdx.z;
  const _Float16* Ab = A + (size_t)bh * sA;
  const _Float16* Bb = B + (size_t)bh * sB;
  float* Cb = C + (size_t)bh * sC;
  const int lane = threadIdx.x & 31;
  const int n0 = blockIdx.x * 32, m0 = blockIdx.y * 32;
  v8f acc[2][2] = {};
  wmma_tile32(Ab, lda, Bb, ldb, K, m0, n0, lane, acc);
  epilogue(m0, n0, lane, acc, [&](int m, int n, float val) {
    Cb[(size_t)m * ldc + n] = val;
  });
}

// ---------------- softmax over 64 (kernel_1 rows), one thread per row ----------------
__global__ void k_softmax64(const float* __restrict__ s, _Float16* __restrict__ o) {
  const int row = blockIdx.x * blockDim.x + threadIdx.x;
  if (row >= BH * N_) return;
  const float* p = s + (size_t)row * LM;
  float mx = -1e30f;
  for (int j = 0; j < LM; ++j) mx = fmaxf(mx, p[j]);
  float sum = 0.f;
  for (int j = 0; j < LM; ++j) sum += __expf(p[j] - mx);
  const float inv = 1.f / sum;
  _Float16* q = o + (size_t)row * LM;
  for (int j = 0; j < LM; ++j) q[j] = (_Float16)(__expf(p[j] - mx) * inv);
}

// ---------------- softmax over 4096 (kernel_3 rows), one block per row ----------------
__global__ void k_softmax4096(const float* __restrict__ s, _Float16* __restrict__ o) {
  __shared__ float red[256];
  const int row = blockIdx.x;           // BH*LM rows
  const float* p = s + (size_t)row * N_;
  _Float16* q = o + (size_t)row * N_;
  const int tid = threadIdx.x;
  float mx = -1e30f;
  for (int j = tid; j < N_; j += 256) mx = fmaxf(mx, p[j]);
  red[tid] = mx; __syncthreads();
  for (int w = 128; w > 0; w >>= 1) { if (tid < w) red[tid] = fmaxf(red[tid], red[tid + w]); __syncthreads(); }
  mx = red[0]; __syncthreads();
  float sum = 0.f;
  for (int j = tid; j < N_; j += 256) sum += __expf(p[j] - mx);
  red[tid] = sum; __syncthreads();
  for (int w = 128; w > 0; w >>= 1) { if (tid < w) red[tid] += red[tid + w]; __syncthreads(); }
  const float inv = 1.f / red[0];
  for (int j = tid; j < N_; j += 256) q[j] = (_Float16)(__expf(p[j] - mx) * inv);
}

// ---------------- 64x64 matmul helper in LDS (256 threads) ----------------
__device__ __forceinline__ void mm64(const float* A, const float* B, float* C, int tid) {
  for (int e = tid; e < 4096; e += 256) {
    const int i = e >> 6, j = e & 63;
    float s = 0.f;
    for (int k = 0; k < 64; ++k) s += A[i * 64 + k] * B[k * 64 + j];
    C[e] = s;
  }
  __syncthreads();
}

// ---------------- kernel_2 softmax + iterative inverse + W2 = inv @ k3v (store W2^T) ----------------
__global__ void k_inv_w2(const _Float16* __restrict__ qlh, const _Float16* __restrict__ klh,
                         const float* __restrict__ k3v, _Float16* __restrict__ w2t) {
  extern __shared__ float sm[];
  float* Km = sm;              // kernel_2 (softmaxed)
  float* Vm = sm + 4096;       // current inverse estimate
  float* KV = sm + 8192;
  float* T1 = sm + 12288;
  float* T2 = sm + 16384;
  __shared__ float red2[64];
  __shared__ float denom_s;
  const int bh = blockIdx.x;
  const int tid = threadIdx.x;  // 256
  const _Float16* qb = qlh + (size_t)bh * LM * HD;
  const _Float16* kb = klh + (size_t)bh * LM * HD;
  // scores = q_land @ k_land^T
  for (int e = tid; e < 4096; e += 256) {
    const int i = e >> 6, j = e & 63;
    float s = 0.f;
    for (int d = 0; d < 64; ++d) s += (float)qb[i * 64 + d] * (float)kb[j * 64 + d];
    Km[e] = s;
  }
  __syncthreads();
  // row softmax
  if (tid < 64) {
    float* rp = Km + tid * 64;
    float mx = -1e30f;
    for (int j = 0; j < 64; ++j) mx = fmaxf(mx, rp[j]);
    float sum = 0.f;
    for (int j = 0; j < 64; ++j) { const float e_ = __expf(rp[j] - mx); rp[j] = e_; sum += e_; }
    const float inv = 1.f / sum;
    for (int j = 0; j < 64; ++j) rp[j] *= inv;
  }
  __syncthreads();
  // V0 = K^T / max_j(colsum_j)
  if (tid < 64) {
    float cs = 0.f;
    for (int i = 0; i < 64; ++i) cs += Km[i * 64 + tid];
    red2[tid] = cs;
  }
  __syncthreads();
  if (tid == 0) {
    float mx = red2[0];
    for (int j = 1; j < 64; ++j) mx = fmaxf(mx, red2[j]);
    denom_s = 1.f / mx;
  }
  __syncthreads();
  for (int e = tid; e < 4096; e += 256) {
    const int i = e >> 6, j = e & 63;
    Vm[e] = Km[j * 64 + i] * denom_s;
  }
  __syncthreads();
  // Newton-Schulz: V = 0.25 V (13I - KV(15I - KV(7I - KV)))
  for (int it = 0; it < 6; ++it) {
    mm64(Km, Vm, KV, tid);
    for (int e = tid; e < 4096; e += 256) { const int i = e >> 6, j = e & 63; T1[e] = (i == j ? 7.f : 0.f) - KV[e]; }
    __syncthreads();
    mm64(KV, T1, T2, tid);
    for (int e = tid; e < 4096; e += 256) { const int i = e >> 6, j = e & 63; T1[e] = (i == j ? 15.f : 0.f) - T2[e]; }
    __syncthreads();
    mm64(KV, T1, T2, tid);
    for (int e = tid; e < 4096; e += 256) { const int i = e >> 6, j = e & 63; T1[e] = (i == j ? 13.f : 0.f) - T2[e]; }
    __syncthreads();
    mm64(Vm, T1, T2, tid);
    for (int e = tid; e < 4096; e += 256) Vm[e] = 0.25f * T2[e];
    __syncthreads();
  }
  // W2 = inv @ k3v, stored transposed f16: w2t[bh][d][l]
  const float* g = k3v + (size_t)bh * 4096;
  for (int e = tid; e < 4096; e += 256) Km[e] = g[e];
  __syncthreads();
  mm64(Vm, Km, T2, tid);
  _Float16* o = w2t + (size_t)bh * 4096;
  for (int e = tid; e < 4096; e += 256) { const int l = e >> 6, d = e & 63; o[d * 64 + l] = (_Float16)T2[e]; }
}

// ---------------- out_heads = kernel_1 @ W2, scatter into [B, N, C] f16 ----------------
__global__ void k_gemm_outheads(const _Float16* __restrict__ k1, const _Float16* __restrict__ w2t,
                                _Float16* __restrict__ attn) {
  const int bh = blockIdx.z;
  const int b = bh / H_;
  const int h = bh % H_;
  const _Float16* A = k1 + (size_t)bh * N_ * LM;
  const _Float16* B = w2t + (size_t)bh * HD * LM;
  const int lane = threadIdx.x & 31;
  const int n0 = blockIdx.x * 32, m0 = blockIdx.y * 32;
  v8f acc[2][2] = {};
  wmma_tile32(A, LM, B, LM, LM, m0, n0, lane, acc);
  epilogue(m0, n0, lane, acc, [&](int m, int n, float val) {
    attn[((size_t)b * N_ + m) * C_ + h * HD + n] = (_Float16)val;
  });
}

// ---------------- final projection: out = attn @ w_proj^T + b_proj ----------------
__global__ void k_gemm_proj(const _Float16* __restrict__ A, const _Float16* __restrict__ W,
                            const float* __restrict__ bias, float* __restrict__ out) {
  const int lane = threadIdx.x & 31;
  const int n0 = blockIdx.x * 32, m0 = blockIdx.y * 32;
  v8f acc[2][2] = {};
  wmma_tile32(A, C_, W, C_, C_, m0, n0, lane, acc);
  epilogue(m0, n0, lane, acc, [&](int m, int n, float val) {
    out[(size_t)m * C_ + n] = val + bias[n];
  });
}

extern "C" void kernel_launch(void* const* d_in, const int* in_sizes, int n_in,
                              void* d_out, int out_size, void* d_ws, size_t ws_size,
                              hipStream_t stream) {
  (void)in_sizes; (void)n_in; (void)out_size; (void)ws_size;
  const float* x      = (const float*)d_in[0];
  const float* w_qkv  = (const float*)d_in[1];
  const float* w_proj = (const float*)d_in[2];
  const float* b_proj = (const float*)d_in[3];
  float* out = (float*)d_out;

  char* ws = (char*)d_ws;
  size_t off = 0;
  auto alloc = [&](size_t bytes) -> char* {
    char* p = ws + off;
    off += (bytes + 255) & ~(size_t)255;
    return p;
  };
  _Float16* xh     = (_Float16*)alloc((size_t)MROWS * C_ * 2);   // 25.2 MB
  _Float16* wqkvh  = (_Float16*)alloc((size_t)C3 * C_ * 2);      //  3.5 MB
  _Float16* wprojh = (_Float16*)alloc((size_t)C_ * C_ * 2);      //  1.2 MB
  _Float16* qh     = (_Float16*)alloc((size_t)BH * N_ * HD * 2); // 25.2 MB
  _Float16* kh     = (_Float16*)alloc((size_t)BH * N_ * HD * 2); // 25.2 MB
  _Float16* vt     = (_Float16*)alloc((size_t)BH * HD * N_ * 2); // 25.2 MB
  _Float16* qlh    = (_Float16*)alloc((size_t)BH * LM * HD * 2);
  _Float16* klh    = (_Float16*)alloc((size_t)BH * LM * HD * 2);
  float*    sc1    = (float*)alloc((size_t)BH * N_ * LM * 4);    // 50.3 MB
  float*    sc3    = (float*)alloc((size_t)BH * LM * N_ * 4);    // 50.3 MB
  float*    k3v    = (float*)alloc((size_t)BH * LM * HD * 4);
  _Float16* w2t    = (_Float16*)alloc((size_t)BH * HD * LM * 2);
  // Aliases (stream order makes these safe):
  _Float16* k1h   = xh;             // xh dead after QKV GEMM; written at step 5
  _Float16* k3h   = (_Float16*)sc1; // sc1 dead after small softmax; written at step 7
  _Float16* attnh = (_Float16*)sc3; // sc3 dead after big softmax; written at step 10

  // 1. f32 -> f16 converts
  { int n = MROWS * C_; k_cvt<<<(n + 255) / 256, 256, 0, stream>>>(x, xh, n); }
  { int n = C3 * C_;    k_cvt<<<(n + 255) / 256, 256, 0, stream>>>(w_qkv, wqkvh, n); }
  { int n = C_ * C_;    k_cvt<<<(n + 255) / 256, 256, 0, stream>>>(w_proj, wprojh, n); }
  // 2. QKV projection + scatter (q scaled by 1/8, v transposed)
  k_gemm_qkv<<<dim3(C3 / 32, MROWS / 32), 32, 0, stream>>>(xh, wqkvh, qh, kh, vt);
  // 3. landmark pooling
  k_pool<<<BH * LM, 64, 0, stream>>>(qh, kh, qlh, klh);
  // 4. kernel_1 scores: q @ k_land^T  -> [bh][4096][64]
  k_gemm_f32out<<<dim3(LM / 32, N_ / 32, BH), 32, 0, stream>>>(
      qh, HD, (size_t)N_ * HD, klh, HD, (size_t)LM * HD, sc1, LM, (size_t)N_ * LM, HD);
  // 5. softmax(64) -> kernel_1 f16
  k_softmax64<<<(BH * N_ + 255) / 256, 256, 0, stream>>>(sc1, k1h);
  // 6. kernel_3 scores: q_land @ k^T  -> [bh][64][4096]
  k_gemm_f32out<<<dim3(N_ / 32, LM / 32, BH), 32, 0, stream>>>(
      qlh, HD, (size_t)LM * HD, kh, HD, (size_t)N_ * HD, sc3, N_, (size_t)LM * N_, HD);
  // 7. softmax(4096) -> kernel_3 f16
  k_softmax4096<<<BH * LM, 256, 0, stream>>>(sc3, k3h);
  // 8. k3v = kernel_3 @ v  (K = 4096) -> [bh][64][64] f32
  k_gemm_f32out<<<dim3(HD / 32, LM / 32, BH), 32, 0, stream>>>(
      k3h, N_, (size_t)LM * N_, vt, N_, (size_t)HD * N_, k3v, HD, (size_t)LM * HD, N_);
  // 9. kernel_2 softmax + Newton inverse + W2^T = (inv @ k3v)^T
  k_inv_w2<<<BH, 256, 5 * 4096 * sizeof(float), stream>>>(qlh, klh, k3v, w2t);
  // 10. out_heads = kernel_1 @ W2 -> attn [B, N, C] f16
  k_gemm_outheads<<<dim3(HD / 32, N_ / 32, BH), 32, 0, stream>>>(k1h, w2t, attnh);
  // 11. out = attn @ w_proj^T + b_proj
  k_gemm_proj<<<dim3(C_ / 32, MROWS / 32), 32, 0, stream>>>(attnh, wprojh, b_proj, out);
}
